// ImageMemory_67473936220402
// MI455X (gfx1250) — matrix operand order repaired and checked by reading, not compiled, not run on hardware
//
#include <hip/hip_runtime.h>

// CDNA5 WMMA fragment types (wave32)
typedef __attribute__((ext_vector_type(16))) __bf16 v16bf;
typedef __attribute__((ext_vector_type(8)))  float  v8f;
typedef __attribute__((ext_vector_type(4)))  float  f32x4;   // native vector for nontemporal builtins

// async-to-LDS builtin pointer types (from hipcc diagnostic: int __vector(4) *)
typedef __attribute__((__vector_size__(16))) int v4i;
typedef __attribute__((address_space(3))) v4i as3_v4i;

#define K_DIM   256
#define M_TOTAL 2048
#define N_TOTAL 100000
#define EPSN    1e-12f

#if __has_builtin(__builtin_amdgcn_global_load_async_to_lds_b128) && \
    __has_builtin(__builtin_amdgcn_s_wait_asynccnt)
#define HAS_ASYNC_LDS 1
#else
#define HAS_ASYNC_LDS 0
#endif

struct B32 { uint4 a, b; };  // 32-byte carrier for a v16bf fragment

__device__ __forceinline__ unsigned short bf16_rne_bits(float f) {
  unsigned int u = __float_as_uint(f);
  u += 0x7fffu + ((u >> 16) & 1u);          // round-to-nearest-even
  return (unsigned short)(u >> 16);
}
__device__ __forceinline__ float bf16_bits_to_f32(unsigned short h) {
  return __uint_as_float(((unsigned int)h) << 16);
}
__device__ __forceinline__ __bf16 bits_to_bf16(unsigned short u) {
  return __builtin_bit_cast(__bf16, u);
}
__device__ __forceinline__ void split_elem(float f, v16bf& hi, v16bf& lo, int e) {
  unsigned short h = bf16_rne_bits(f);
  hi[e] = bits_to_bf16(h);
  lo[e] = bits_to_bf16(bf16_rne_bits(f - bf16_bits_to_f32(h)));
}
__device__ __forceinline__ void split4s(float4 f, float s, v16bf& hi, v16bf& lo, int base) {
  split_elem(f.x * s, hi, lo, base + 0);
  split_elem(f.y * s, hi, lo, base + 1);
  split_elem(f.z * s, hi, lo, base + 2);
  split_elem(f.w * s, hi, lo, base + 3);
}
__device__ __forceinline__ v16bf load_frag(const unsigned short* p) {
  B32 q;
  q.a = ((const uint4*)p)[0];
  q.b = ((const uint4*)p)[1];
  return __builtin_bit_cast(v16bf, q);
}
// generic LDS pointer -> 32-bit LDS offset pointer (ISA 10.2: LDS address is the low 32 bits)
__device__ __forceinline__ as3_v4i* to_lds_v4i(void* p) {
  return (as3_v4i*)(unsigned int)(unsigned long long)p;
}

// ---------------------------------------------------------------------------
// Kernel 1: per-row L2 norms. One wave32 per row, 8 rows per 256-thread block.
// ---------------------------------------------------------------------------
__global__ void __launch_bounds__(256)
row_norm_kernel(const float* __restrict__ x, float* __restrict__ norms) {
  const int wave = threadIdx.x >> 5;
  const int lane = threadIdx.x & 31;
  const int row  = blockIdx.x * 8 + wave;
  const float* xr = x + (size_t)row * K_DIM;
  float s = 0.0f;
#pragma unroll
  for (int i = 0; i < 8; ++i) { float v = xr[lane + 32 * i]; s += v * v; }
#pragma unroll
  for (int m = 16; m >= 1; m >>= 1) s += __shfl_xor(s, m, 32);
  if (lane == 0) norms[row] = fmaxf(sqrtf(s), EPSN);
}

// ---------------------------------------------------------------------------
// Kernel 2: normalize x, split bf16 hi/lo, pack as WMMA 16x16x32 A operand, contiguous k order:
//   lane = M%16 + 16*khalf ; element e -> K = 16*khalf + e
// ---------------------------------------------------------------------------
__global__ void __launch_bounds__(32)
pack_a_kernel(const float* __restrict__ x, const float* __restrict__ norms,
              unsigned short* __restrict__ Ahi, unsigned short* __restrict__ Alo) {
  const int tile  = blockIdx.x;          // tile = mtile*8 + ktile
  const int ktile = tile & 7;
  const int lane  = threadIdx.x & 31;
  const int half  = lane >> 4;
  const int mrow  = ((tile >> 3) << 4) + (lane & 15);
  const float inv = 1.0f / norms[mrow];

  const float* xr = x + (size_t)mrow * K_DIM + ktile * 32 + half * 16;
  float4 f0 = ((const float4*)xr)[0];
  float4 f1 = ((const float4*)xr)[1];
  float4 f2 = ((const float4*)(xr + 8))[0];
  float4 f3 = ((const float4*)(xr + 8))[1];

  v16bf hi, lo;
  split4s(f0, inv, hi, lo, 0);
  split4s(f1, inv, hi, lo, 4);
  split4s(f2, inv, hi, lo, 8);
  split4s(f3, inv, hi, lo, 12);

  B32 hb = __builtin_bit_cast(B32, hi);
  B32 lb = __builtin_bit_cast(B32, lo);
  uint4* ph = (uint4*)(Ahi + ((size_t)tile * 32 + lane) * 16);
  uint4* pl = (uint4*)(Alo + ((size_t)tile * 32 + lane) * 16);
  ph[0] = hb.a; ph[1] = hb.b;
  pl[0] = lb.a; pl[1] = lb.b;
}

// ---------------------------------------------------------------------------
// Kernel 3: one-shot split of features into bf16 hi/lo, packed in the WMMA
// B-matrix operand, same contiguous k order as A (lane = N%16 + 16*khalf, element e -> K = 16*khalf + e).
// Same byte footprint as f32 features (2x 51.2 MB) -> stays L2 resident.
// ---------------------------------------------------------------------------
__global__ void __launch_bounds__(256)
pack_b_kernel(const float* __restrict__ feat,
              unsigned short* __restrict__ Bhi, unsigned short* __restrict__ Blo) {
  const int wave = threadIdx.x >> 5;
  const int lane = threadIdx.x & 31;
  const int tile = blockIdx.x * 8 + wave;     // tile = ntile*8 + ktile
  const int ktile = tile & 7;
  const int ncol  = ((tile >> 3) << 4) + (lane & 15);
  const int half  = lane >> 4;

  const float* fp = feat + (size_t)ncol * K_DIM + ktile * 32 + half * 16;
  float4 f0 = ((const float4*)fp)[0];
  float4 f1 = ((const float4*)fp)[1];
  float4 f2 = ((const float4*)fp)[2];
  float4 f3 = ((const float4*)fp)[3];

  v16bf hi, lo;
  split4s(f0, 1.0f, hi, lo, 0);
  split4s(f1, 1.0f, hi, lo, 4);
  split4s(f2, 1.0f, hi, lo, 8);
  split4s(f3, 1.0f, hi, lo, 12);

  B32 hb = __builtin_bit_cast(B32, hi);
  B32 lb = __builtin_bit_cast(B32, lo);
  uint4* ph = (uint4*)(Bhi + ((size_t)tile * 32 + lane) * 16);
  uint4* pl = (uint4*)(Blo + ((size_t)tile * 32 + lane) * 16);
  ph[0] = hb.a; ph[1] = hb.b;
  pl[0] = lb.a; pl[1] = lb.b;
}

// ---------------------------------------------------------------------------
// Kernel 4 (primary): bf16x3 GEMM from pre-packed fragments.
// Block = 256 threads = 8 waves; wave owns two 16-row M-tiles (TM=2), block
// shares a 64-col N strip (TN=4). B fragments staged into double-buffered LDS
// (2 x 8 KB) with GLOBAL_LOAD_ASYNC_TO_LDS_B128 + s_wait_asynccnt: next
// K-step's B strip streams into LDS while WMMAs consume the current one.
// ---------------------------------------------------------------------------
__global__ void __launch_bounds__(256)
wmma_gemm_packed(const unsigned short* __restrict__ Ahi,
                 const unsigned short* __restrict__ Alo,
                 const unsigned short* __restrict__ Bhi,
                 const unsigned short* __restrict__ Blo,
                 float* __restrict__ out) {
  __shared__ unsigned short sB[2][4][2][32][16];   // [buf][subtile][hi/lo][lane][16] = 16 KB

  const int tid  = threadIdx.x;
  const int wave = tid >> 5;
  const int lane = tid & 31;
  const int half = lane >> 4;
  const int lcol = lane & 15;
  const int mt0  = (blockIdx.y * 8 + wave) * 2;   // two M-tiles: mt0, mt0+1
  const int nt0  = blockIdx.x * 4;                // four N-subtiles

  // Per-thread staging chunks: c = tid + r*256, r in {0,1}; 512 x 16B = 8 KB.
  // Chunk c -> subtile s = c>>7, h = (c>>6)&1, rem = c&63; flat layouts match.
  const int c0 = tid, c1 = tid + 256;
  const int s0 = c0 >> 7, h0 = (c0 >> 6) & 1, r0 = c0 & 63;
  const int s1 = c1 >> 7, h1 = (c1 >> 6) & 1, r1 = c1 & 63;
  const bool ok0 = (nt0 + s0) * 16 < N_TOTAL;
  const bool ok1 = (nt0 + s1) * 16 < N_TOTAL;
  const unsigned short* base0 = (h0 ? Blo : Bhi) + ((size_t)(nt0 + s0) * 8) * 512 + r0 * 8;
  const unsigned short* base1 = (h1 ? Blo : Bhi) + ((size_t)(nt0 + s1) * 8) * 512 + r1 * 8;
  // per-kt source advance: +512 ushorts (one (ntile,kt) fragment)

  v8f acc[2][4] = {};

#if HAS_ASYNC_LDS
  // Issue kt=0 stage into buffer 0 (per-lane 16B async transfers, EXEC-masked).
  {
    as3_v4i* d0 = to_lds_v4i((uint4*)&sB[0][0][0][0][0] + c0);
    as3_v4i* d1 = to_lds_v4i((uint4*)&sB[0][0][0][0][0] + c1);
    if (ok0) __builtin_amdgcn_global_load_async_to_lds_b128((v4i*)base0, d0, 0, 0);
    if (ok1) __builtin_amdgcn_global_load_async_to_lds_b128((v4i*)base1, d1, 0, 0);
  }
#pragma unroll
  for (int kt = 0; kt < 8; ++kt) {
    const int buf = kt & 1;
    __builtin_amdgcn_s_wait_asynccnt(0);   // this wave's async stages landed
    __syncthreads();                        // everyone's stages landed
    if (kt < 7) {                           // stream next K-step into other buffer
      as3_v4i* d0 = to_lds_v4i((uint4*)&sB[buf ^ 1][0][0][0][0] + c0);
      as3_v4i* d1 = to_lds_v4i((uint4*)&sB[buf ^ 1][0][0][0][0] + c1);
      if (ok0) __builtin_amdgcn_global_load_async_to_lds_b128(
                   (v4i*)(base0 + (size_t)(kt + 1) * 512), d0, 0, 0);
      if (ok1) __builtin_amdgcn_global_load_async_to_lds_b128(
                   (v4i*)(base1 + (size_t)(kt + 1) * 512), d1, 0, 0);
    }
#else
  // Register-prefetch pipeline: global loads for kt+1 issued during kt compute.
  uint4 p0 = {}, p1 = {};
  if (ok0) p0 = *(const uint4*)base0;
  if (ok1) p1 = *(const uint4*)base1;
#pragma unroll
  for (int kt = 0; kt < 8; ++kt) {
    const int buf = kt & 1;
    if (kt > 0) __syncthreads();            // previous buffer's reads done
    if (ok0) ((uint4*)&sB[buf][0][0][0][0])[c0] = p0;
    if (ok1) ((uint4*)&sB[buf][0][0][0][0])[c1] = p1;
    __syncthreads();
    if (kt < 7) {
      if (ok0) p0 = *(const uint4*)(base0 + (size_t)(kt + 1) * 512);
      if (ok1) p1 = *(const uint4*)(base1 + (size_t)(kt + 1) * 512);
    }
#endif

    // ---- A fragments for this wave's two M-tiles ----
    const size_t a0 = (((size_t)mt0 * 8 + kt) * 32 + lane) * 16;
    const size_t a1 = ((((size_t)mt0 + 1) * 8 + kt) * 32 + lane) * 16;
    const v16bf a0h = load_frag(Ahi + a0);
    const v16bf a0l = load_frag(Alo + a0);
    const v16bf a1h = load_frag(Ahi + a1);
    const v16bf a1l = load_frag(Alo + a1);

#pragma unroll
    for (int s = 0; s < 4; ++s) {
      if ((nt0 + s) * 16 < N_TOTAL) {      // wave-uniform: EXEC stays all-ones
        const v16bf bh = load_frag(&sB[buf][s][0][lane][0]);
        const v16bf bl = load_frag(&sB[buf][s][1][lane][0]);
        // bf16x3: xh*fh + xl*fh + xh*fl
        acc[0][s] = __builtin_amdgcn_wmma_f32_16x16x32_bf16(false, a0h, false, bh, (short)0, acc[0][s], false, false);
        acc[0][s] = __builtin_amdgcn_wmma_f32_16x16x32_bf16(false, a0l, false, bh, (short)0, acc[0][s], false, false);
        acc[0][s] = __builtin_amdgcn_wmma_f32_16x16x32_bf16(false, a0h, false, bl, (short)0, acc[0][s], false, false);
        acc[1][s] = __builtin_amdgcn_wmma_f32_16x16x32_bf16(false, a1h, false, bh, (short)0, acc[1][s], false, false);
        acc[1][s] = __builtin_amdgcn_wmma_f32_16x16x32_bf16(false, a1l, false, bh, (short)0, acc[1][s], false, false);
        acc[1][s] = __builtin_amdgcn_wmma_f32_16x16x32_bf16(false, a1h, false, bl, (short)0, acc[1][s], false, false);
      }
    }
  }

  // C/D layout: element v -> row M = mtile*16 + v + 8*half, col = n0 + lcol
#pragma unroll
  for (int m = 0; m < 2; ++m) {
#pragma unroll
    for (int s = 0; s < 4; ++s) {
      const int n0 = (nt0 + s) * 16;
      if (n0 < N_TOTAL) {
        float* op = out + (size_t)((mt0 + m) * 16 + 8 * half) * N_TOTAL + (n0 + lcol);
#pragma unroll
        for (int v = 0; v < 8; ++v)
          __builtin_nontemporal_store(acc[m][s][v], op + (size_t)v * N_TOTAL);
      }
    }
  }
}

// ---------------------------------------------------------------------------
// Kernel 4b (fallback, small ws): on-the-fly conversion GEMM.
// ---------------------------------------------------------------------------
__global__ void __launch_bounds__(256)
wmma_gemm_fly(const float* __restrict__ feat,
              const unsigned short* __restrict__ Ahi,
              const unsigned short* __restrict__ Alo,
              float* __restrict__ out) {
  const int wave  = threadIdx.x >> 5;
  const int lane  = threadIdx.x & 31;
  const int half  = lane >> 4;
  const int lcol  = lane & 15;
  const int mtile = blockIdx.y * 8 + wave;
  const int nbase = blockIdx.x * 64;

  v8f acc[4] = {};

  for (int kt = 0; kt < 8; ++kt) {
    const size_t abase = (((size_t)mtile * 8 + kt) * 32 + lane) * 16;
    const v16bf a_hi = load_frag(Ahi + abase);
    const v16bf a_lo = load_frag(Alo + abase);
    const int krow = kt * 32 + half * 16;
#pragma unroll
    for (int s = 0; s < 4; ++s) {
      const int n0 = nbase + s * 16;
      if (n0 < N_TOTAL) {
        const float* fp = feat + (size_t)(n0 + lcol) * K_DIM + krow;
        float4 f0 = ((const float4*)fp)[0];
        float4 f1 = ((const float4*)fp)[1];
        float4 f2 = ((const float4*)fp)[2];
        float4 f3 = ((const float4*)fp)[3];
        v16bf b_hi, b_lo;
        split4s(f0, 1.0f, b_hi, b_lo, 0);
        split4s(f1, 1.0f, b_hi, b_lo, 4);
        split4s(f2, 1.0f, b_hi, b_lo, 8);
        split4s(f3, 1.0f, b_hi, b_lo, 12);
        acc[s] = __builtin_amdgcn_wmma_f32_16x16x32_bf16(false, a_hi, false, b_hi, (short)0, acc[s], false, false);
        acc[s] = __builtin_amdgcn_wmma_f32_16x16x32_bf16(false, a_lo, false, b_hi, (short)0, acc[s], false, false);
        acc[s] = __builtin_amdgcn_wmma_f32_16x16x32_bf16(false, a_hi, false, b_lo, (short)0, acc[s], false, false);
      }
    }
  }
#pragma unroll
  for (int s = 0; s < 4; ++s) {
    const int n0 = nbase + s * 16;
    if (n0 < N_TOTAL) {
      float* op = out + (size_t)(mtile * 16 + 8 * half) * N_TOTAL + (n0 + lcol);
#pragma unroll
      for (int v = 0; v < 8; ++v)
        __builtin_nontemporal_store(acc[s][v], op + (size_t)v * N_TOTAL);
    }
  }
}

// ---------------------------------------------------------------------------
// Kernel 5: second tuple output = features passthrough (non-temporal copy).
// ---------------------------------------------------------------------------
__global__ void __launch_bounds__(256)
copy_feat_kernel(const f32x4* __restrict__ src, f32x4* __restrict__ dst, int n4) {
  int i = blockIdx.x * blockDim.x + threadIdx.x;
  if (i < n4) {
    f32x4 v = __builtin_nontemporal_load(src + i);
    __builtin_nontemporal_store(v, dst + i);
  }
}

extern "C" void kernel_launch(void* const* d_in, const int* in_sizes, int n_in,
                              void* d_out, int out_size, void* d_ws, size_t ws_size,
                              hipStream_t stream) {
  (void)in_sizes; (void)n_in; (void)out_size;
  const float* x    = (const float*)d_in[0];   // [2048, 256]
  // d_in[1] = targets (unused by the reference outputs)
  const float* feat = (const float*)d_in[2];   // [100000, 256]
  float* out = (float*)d_out;

  // Workspace layout: norms(8KB) | Ahi(1MB) | Alo(1MB) | Bhi(51.2MB) | Blo(51.2MB)
  const size_t A_BYTES = (size_t)M_TOTAL * K_DIM * 2;
  const size_t B_BYTES = (size_t)N_TOTAL * K_DIM * 2;
  float*          norms = (float*)d_ws;
  unsigned short* Ahi   = (unsigned short*)((char*)d_ws + 8192);
  unsigned short* Alo   = (unsigned short*)((char*)d_ws + 8192 + A_BYTES);
  unsigned short* Bhi   = (unsigned short*)((char*)d_ws + 8192 + 2 * A_BYTES);
  unsigned short* Blo   = (unsigned short*)((char*)d_ws + 8192 + 2 * A_BYTES + B_BYTES);
  const size_t need = 8192 + 2 * A_BYTES + 2 * B_BYTES;   // ~104.5 MB

  row_norm_kernel<<<M_TOTAL / 8, 256, 0, stream>>>(x, norms);
  pack_a_kernel<<<(M_TOTAL / 16) * (K_DIM / 32), 32, 0, stream>>>(x, norms, Ahi, Alo);

  if (ws_size >= need) {
    pack_b_kernel<<<(N_TOTAL / 16) * (K_DIM / 32) / 8, 256, 0, stream>>>(feat, Bhi, Blo);
    dim3 grid((N_TOTAL + 63) / 64, M_TOTAL / 256);   // 1563 x 8
    wmma_gemm_packed<<<grid, 256, 0, stream>>>(Ahi, Alo, Bhi, Blo, out);
  } else {
    dim3 grid((N_TOTAL + 63) / 64, M_TOTAL / 128);   // 1563 x 16
    wmma_gemm_fly<<<grid, 256, 0, stream>>>(feat, Ahi, Alo, out);
  }

  const int n4 = (N_TOTAL * K_DIM) / 4;
  copy_feat_kernel<<<(n4 + 255) / 256, 256, 0, stream>>>(
      (const f32x4*)feat, (f32x4*)(out + (size_t)M_TOTAL * N_TOTAL), n4);
}
